// PPO_45595372814995
// MI455X (gfx1250) — matrix operand order repaired and checked by reading, not compiled, not run on hardware
//
#include <hip/hip_runtime.h>
#include <hip/hip_bf16.h>

// ---------------------------------------------------------------------------
// PPO loss, fused for gfx1250 (MI455X).
//   T=131072, D=512, A=64, H=128. GEMMs done with v_wmma_f32_16x16x32_bf16
//   using bf16 hi/lo splitting (3 WMMAs per tile-step) for ~fp32 accuracy.
// ---------------------------------------------------------------------------

typedef __attribute__((ext_vector_type(16))) __bf16 v16bf;
typedef __attribute__((ext_vector_type(8)))  float  v8f;

#define T_TOT   131072
#define D_IN    512
#define N_ACT   64
#define H_HID   128
#define GAMMA_F 0.99f
#define LMBDA_F 0.95f

// ---- workspace layout (bytes) ---------------------------------------------
// packed weights: [ntile][kstep][lane(32)][e(16)] bf16, hi and lo parts
#define OFF_WC_HI 0            // critic_w1: 8*16*32*16 = 65536 bf16 = 131072 B
#define OFF_WC_LO 131072
#define OFF_WA_HI 262144       // actor_w1
#define OFF_WA_LO 393216
#define OFF_W3_HI 524288       // actor_w3: 4*4*32*16 = 8192 bf16 = 16384 B
#define OFF_W3_LO 540672
#define OFF_DELTA 557056       // T f32
#define OFF_RATIO 1081344      // T f32
#define OFF_ENT   1605632      // T f32

// ===========================================================================
// Kernel 1: pack weights into WMMA B-fragment order, split into bf16 hi/lo.
//   B fragment (32x16 bf16): lane -> column n = lane%16; element e -> K =
//   (e&7) + 8*(lane/16) + 16*(e/8). Packed so each lane's 16 bf16 are contiguous (32B).
// ===========================================================================
__global__ void ppo_prep_pack(const float* __restrict__ cw1,
                              const float* __restrict__ aw1,
                              const float* __restrict__ aw3,
                              __bf16* __restrict__ wcHi, __bf16* __restrict__ wcLo,
                              __bf16* __restrict__ waHi, __bf16* __restrict__ waLo,
                              __bf16* __restrict__ w3Hi, __bf16* __restrict__ w3Lo)
{
    const int total = 65536 + 65536 + 8192;
    for (int i = blockIdx.x * blockDim.x + threadIdx.x; i < total;
         i += gridDim.x * blockDim.x) {
        const float* src;
        __bf16 *dhi, *dlo;
        int idx, ncols, tile, ks;
        if (i < 65536)       { src = cw1; dhi = wcHi; dlo = wcLo; idx = i;           ncols = 128; }
        else if (i < 131072) { src = aw1; dhi = waHi; dlo = waLo; idx = i - 65536;   ncols = 128; }
        else                 { src = aw3; dhi = w3Hi; dlo = w3Lo; idx = i - 131072;  ncols = 64;  }
        int e    = idx & 15;
        int lane = (idx >> 4) & 31;
        int rest = idx >> 9;
        if (ncols == 128) { ks = rest & 15; tile = rest >> 4; }   // 16 ksteps, 8 tiles
        else              { ks = rest & 3;  tile = rest >> 2; }   // 4 ksteps, 4 tiles
        int n = tile * 16 + (lane & 15);
        int K = ks * 32 + ((e & 8) << 1) + ((lane >> 4) << 3) + (e & 7);
        float x  = src[K * ncols + n];
        __bf16 h = (__bf16)x;
        __bf16 l = (__bf16)(x - (float)h);
        dhi[idx] = h;
        dlo[idx] = l;
    }
}

// ---------------------------------------------------------------------------
// Convert a 16x512 f32 tile into A-fragment-ordered bf16 hi/lo in LDS.
// 16-bit A 16x32 layout (ISA 7.12.2): lane = row + 16*half,
//   half0: e-> K = {0..7} U {16..23}; half1: K = {8..15} U {24..31}.
// ---------------------------------------------------------------------------
__device__ inline void convert_tile(const float* __restrict__ src,
                                    __bf16 (*aHi)[32][16],
                                    __bf16 (*aLo)[32][16], int tid)
{
    for (int idx = tid; idx < 16 * D_IN; idx += 256) {
        int row = idx >> 9;          // 0..15
        int K   = idx & 511;         // 0..511
        float x  = src[(long)row * D_IN + K];
        __bf16 h = (__bf16)x;
        __bf16 l = (__bf16)(x - (float)h);
        int ksd = K >> 5, kk = K & 31;
        int half, v;
        if      (kk < 8)  { half = 0; v = kk >> 1; }
        else if (kk < 16) { half = 1; v = (kk - 8) >> 1; }
        else if (kk < 24) { half = 0; v = 4 + ((kk - 16) >> 1); }
        else              { half = 1; v = 4 + ((kk - 24) >> 1); }
        int e = (v << 1) | (kk & 1);
        aHi[ksd][row + 16 * half][e] = h;
        aLo[ksd][row + 16 * half][e] = l;
    }
}

// ===========================================================================
// Kernel 2: fused per-16-row-tile pipeline.
// ===========================================================================
__global__ void __launch_bounds__(256)
ppo_fused(const float* __restrict__ s, const float* __restrict__ sp,
          const float* __restrict__ rw, const unsigned char* __restrict__ done,
          const int* __restrict__ act, const float* __restrict__ lpa_old,
          const float* __restrict__ ab1, const float* __restrict__ ab3,
          const float* __restrict__ cb1, const float* __restrict__ cwv,
          const float* __restrict__ cbv,
          const __bf16* __restrict__ wcHi, const __bf16* __restrict__ wcLo,
          const __bf16* __restrict__ waHi, const __bf16* __restrict__ waLo,
          const __bf16* __restrict__ w3Hi, const __bf16* __restrict__ w3Lo,
          float* __restrict__ delta, float* __restrict__ ratio,
          float* __restrict__ ent)
{
    __shared__ __bf16 aHi[16][32][16];     // 16 KB  A fragments (hi)
    __shared__ __bf16 aLo[16][32][16];     // 16 KB  A fragments (lo)
    __shared__ float  hid[16][264];        // 16.9 KB hidden (critic 0..127 | actor 128..255)
    __shared__ float  lg[16][64];          // 4 KB   actor logits
    __shared__ float  vS[16];

    const int tid  = threadIdx.x;
    const int lane = tid & 31;
    const int wave = tid >> 5;
    const long row0 = (long)blockIdx.x * 16;

    // ---- phase 1: stage + split s tile ------------------------------------
    convert_tile(s + row0 * D_IN, aHi, aLo, tid);
    __syncthreads();

    // ---- phase 2: first layer on s (16 col-tile jobs: 8 critic + 8 actor) -
    for (int rep = 0; rep < 2; ++rep) {
        const int  job     = wave + 8 * rep;
        const bool isActor = job >= 8;
        const int  tile    = isActor ? job - 8 : job;
        const __bf16* bhB  = (isActor ? waHi : wcHi) + (tile << 13);
        const __bf16* blB  = (isActor ? waLo : wcLo) + (tile << 13);
        v8f acc = {0.f, 0.f, 0.f, 0.f, 0.f, 0.f, 0.f, 0.f};
#pragma unroll
        for (int ks = 0; ks < 16; ++ks) {
            v16bf ah = *(const v16bf*)(&aHi[ks][lane][0]);
            v16bf al = *(const v16bf*)(&aLo[ks][lane][0]);
            v16bf bh = *(const v16bf*)(bhB + (((ks << 5) + lane) << 4));
            v16bf bl = *(const v16bf*)(blB + (((ks << 5) + lane) << 4));
            acc = __builtin_amdgcn_wmma_f32_16x16x32_bf16(false, ah, false, bh, (short)0, acc, false, false);
            acc = __builtin_amdgcn_wmma_f32_16x16x32_bf16(false, ah, false, bl, (short)0, acc, false, false);
            acc = __builtin_amdgcn_wmma_f32_16x16x32_bf16(false, al, false, bh, (short)0, acc, false, false);
        }
        const int half = lane >> 4, n = lane & 15;
        const int colg = tile * 16 + n;
        const float bias = isActor ? ab1[colg] : cb1[colg];
        const int col = (isActor ? 128 : 0) + colg;
#pragma unroll
        for (int v = 0; v < 8; ++v) {
            float x = acc[v] + bias;
            hid[v + 8 * half][col] = x > 0.f ? x : 0.f;   // ReLU
        }
    }
    __syncthreads();

    // ---- phase 3: actor layer 2 (waves 0..3) + critic value dot (tid<16) --
    if (wave < 4) {
        const int tile = wave;
        v8f acc = {0.f, 0.f, 0.f, 0.f, 0.f, 0.f, 0.f, 0.f};
        const int half = lane >> 4, rowA = lane & 15;
#pragma unroll
        for (int ks = 0; ks < 4; ++ks) {
            v16bf ah, al;
#pragma unroll
            for (int e = 0; e < 16; ++e) {
                const int v = e >> 1, pos = e & 1;
                const int kk = (v < 4 ? 2 * v : 8 + 2 * v) + pos + 8 * half;
                float x  = hid[rowA][128 + ks * 32 + kk];
                __bf16 h = (__bf16)x;
                ah[e] = h;
                al[e] = (__bf16)(x - (float)h);
            }
            v16bf bh = *(const v16bf*)(w3Hi + ((((tile << 2) + ks) * 32 + lane) << 4));
            v16bf bl = *(const v16bf*)(w3Lo + ((((tile << 2) + ks) * 32 + lane) << 4));
            acc = __builtin_amdgcn_wmma_f32_16x16x32_bf16(false, ah, false, bh, (short)0, acc, false, false);
            acc = __builtin_amdgcn_wmma_f32_16x16x32_bf16(false, ah, false, bl, (short)0, acc, false, false);
            acc = __builtin_amdgcn_wmma_f32_16x16x32_bf16(false, al, false, bh, (short)0, acc, false, false);
        }
        const int n = lane & 15;
        const float b = ab3[tile * 16 + n];
#pragma unroll
        for (int v = 0; v < 8; ++v)
            lg[v + 8 * half][tile * 16 + n] = acc[v] + b;
    }
    if (tid < 16) {                       // v_s = relu-hidden . wv + bv
        float accv = cbv[0];
        for (int k = 0; k < H_HID; ++k) accv += hid[tid][k] * cwv[k];
        vS[tid] = accv;
    }
    __syncthreads();

    // ---- phase 4: per-row softmax stats; others begin staging s' ----------
    if (tid < 16) {
        float m = -3.4e38f;
        for (int c = 0; c < N_ACT; ++c) m = fmaxf(m, lg[tid][c]);
        float se = 0.f, sz = 0.f;
        for (int c = 0; c < N_ACT; ++c) {
            float z  = lg[tid][c];
            float ez = __expf(z - m);
            se += ez;
            sz += ez * z;
        }
        float lse     = m + __logf(se);
        float entropy = lse - sz / se;        // = -sum(pi*logp)
        long  t  = row0 + tid;
        int   ai = act[t];
        float log_pi_a = lg[tid][ai] - lse;
        ratio[t] = __expf(log_pi_a - lpa_old[t]);
        ent[t]   = entropy;
    }
    convert_tile(sp + row0 * D_IN, aHi, aLo, tid);   // stage s' (safe: aPack idle)
    __syncthreads();

    // ---- phase 5: first layer (critic only) on s' -------------------------
    {
        const int tile = wave;                        // 8 jobs / 8 waves
        const __bf16* bhB = wcHi + (tile << 13);
        const __bf16* blB = wcLo + (tile << 13);
        v8f acc = {0.f, 0.f, 0.f, 0.f, 0.f, 0.f, 0.f, 0.f};
#pragma unroll
        for (int ks = 0; ks < 16; ++ks) {
            v16bf ah = *(const v16bf*)(&aHi[ks][lane][0]);
            v16bf al = *(const v16bf*)(&aLo[ks][lane][0]);
            v16bf bh = *(const v16bf*)(bhB + (((ks << 5) + lane) << 4));
            v16bf bl = *(const v16bf*)(blB + (((ks << 5) + lane) << 4));
            acc = __builtin_amdgcn_wmma_f32_16x16x32_bf16(false, ah, false, bh, (short)0, acc, false, false);
            acc = __builtin_amdgcn_wmma_f32_16x16x32_bf16(false, ah, false, bl, (short)0, acc, false, false);
            acc = __builtin_amdgcn_wmma_f32_16x16x32_bf16(false, al, false, bh, (short)0, acc, false, false);
        }
        const int half = lane >> 4, n = lane & 15;
        const int colg = tile * 16 + n;
        const float bias = cb1[colg];
#pragma unroll
        for (int v = 0; v < 8; ++v) {
            float x = acc[v] + bias;
            hid[v + 8 * half][colg] = x > 0.f ? x : 0.f;
        }
    }
    __syncthreads();

    // ---- phase 6: v_s', td_target, delta ----------------------------------
    if (tid < 16) {
        float accv = cbv[0];
        for (int k = 0; k < H_HID; ++k) accv += hid[tid][k] * cwv[k];
        long  t  = row0 + tid;
        float dm = done[t] ? 1.f : 0.f;               // jax bool = 1 byte
        float td = rw[t] + GAMMA_F * accv * dm;
        delta[t] = td - vS[tid];
    }
}

// ===========================================================================
// Kernel 3: GAE reverse scan (chunked suffix scan) + loss reduction.
//   adv[t] = delta[t] + C*adv[t+1], C = GAMMA*LMBDA.
// ===========================================================================
__global__ void __launch_bounds__(1024)
ppo_gae_loss(const float* __restrict__ delta, const float* __restrict__ ratio,
             const float* __restrict__ ent, float* __restrict__ out)
{
    const int   NT = 1024, L = T_TOT / 1024;          // L = 128
    const float C  = GAMMA_F * LMBDA_F;
    __shared__ float sS[1024];
    __shared__ float red[1024];
    __shared__ float totals[3];

    const int  tid = threadIdx.x;
    const long t0  = (long)tid * L;

    // local aggregate S = sum_{i<L} C^i * delta[t0+i]  (= adv at chunk head
    // assuming zero incoming carry)
    float S = 0.f, cp = 1.f;
    for (int i = 0; i < L; ++i) { S = fmaf(cp, delta[t0 + i], S); cp *= C; }
    const float cL = cp;                              // C^L

    // inclusive suffix scan over chunk aggregates: sS[j] <- sum_{k>=j} cL^(k-j) S_k
    sS[tid] = S;
    __syncthreads();
    float val = S, p = cL;
    for (int d = 1; d < NT; d <<= 1) {
        float other = (tid + d < NT) ? sS[tid + d] : 0.f;
        __syncthreads();
        val += p * other;
        sS[tid] = val;
        p *= p;
        __syncthreads();
    }
    const float carry = (tid + 1 < NT) ? sS[tid + 1] : 0.f;  // adv at chunk end+1

    // replay chunk backwards, accumulating loss terms
    float accP = 0.f, accV = 0.f, accE = 0.f;
    float adv = carry;
    for (int i = L - 1; i >= 0; --i) {
        const long t = t0 + i;
        const float d = delta[t];
        adv = fmaf(C, adv, d);
        const float rt = ratio[t];
        const float rc = fminf(fmaxf(rt, 0.8f), 1.2f);
        accP -= fminf(rt * adv, rc * adv);
        const float ad = fabsf(d);
        accV += (ad < 1.f) ? 0.5f * d * d : (ad - 0.5f);
        accE += ent[t];
    }

    // tree reductions (deterministic)
    red[tid] = accP; __syncthreads();
    for (int s2 = 512; s2 > 0; s2 >>= 1) { if (tid < s2) red[tid] += red[tid + s2]; __syncthreads(); }
    if (tid == 0) totals[0] = red[0];
    __syncthreads();
    red[tid] = accV; __syncthreads();
    for (int s2 = 512; s2 > 0; s2 >>= 1) { if (tid < s2) red[tid] += red[tid + s2]; __syncthreads(); }
    if (tid == 0) totals[1] = red[0];
    __syncthreads();
    red[tid] = accE; __syncthreads();
    for (int s2 = 512; s2 > 0; s2 >>= 1) { if (tid < s2) red[tid] += red[tid + s2]; __syncthreads(); }
    if (tid == 0) {
        const float invT = 1.f / (float)T_TOT;
        out[0] = totals[0] * invT + 0.5f * (totals[1] * invT) - 0.01f * (red[0] * invT);
    }
}

// ===========================================================================
extern "C" void kernel_launch(void* const* d_in, const int* in_sizes, int n_in,
                              void* d_out, int out_size, void* d_ws, size_t ws_size,
                              hipStream_t stream)
{
    (void)in_sizes; (void)n_in; (void)out_size; (void)ws_size;
    const float*         s    = (const float*)d_in[0];
    const float*         sp   = (const float*)d_in[1];
    const float*         rw   = (const float*)d_in[2];
    const unsigned char* done = (const unsigned char*)d_in[3];   // jax bool, 1B
    const int*           act  = (const int*)d_in[4];
    const float*         lpa  = (const float*)d_in[5];
    const float* aw1 = (const float*)d_in[6];
    const float* ab1 = (const float*)d_in[7];
    const float* aw3 = (const float*)d_in[8];
    const float* ab3 = (const float*)d_in[9];
    const float* cw1 = (const float*)d_in[10];
    const float* cb1 = (const float*)d_in[11];
    const float* cwv = (const float*)d_in[12];
    const float* cbv = (const float*)d_in[13];

    char* ws = (char*)d_ws;
    __bf16* wcHi = (__bf16*)(ws + OFF_WC_HI);
    __bf16* wcLo = (__bf16*)(ws + OFF_WC_LO);
    __bf16* waHi = (__bf16*)(ws + OFF_WA_HI);
    __bf16* waLo = (__bf16*)(ws + OFF_WA_LO);
    __bf16* w3Hi = (__bf16*)(ws + OFF_W3_HI);
    __bf16* w3Lo = (__bf16*)(ws + OFF_W3_LO);
    float*  dlt  = (float*)(ws + OFF_DELTA);
    float*  rto  = (float*)(ws + OFF_RATIO);
    float*  en   = (float*)(ws + OFF_ENT);

    ppo_prep_pack<<<544, 256, 0, stream>>>(cw1, aw1, aw3,
                                           wcHi, wcLo, waHi, waLo, w3Hi, w3Lo);
    ppo_fused<<<T_TOT / 16, 256, 0, stream>>>(s, sp, rw, done, act, lpa,
                                              ab1, ab3, cb1, cwv, cbv,
                                              wcHi, wcLo, waHi, waLo, w3Hi, w3Lo,
                                              dlt, rto, en);
    ppo_gae_loss<<<1, 1024, 0, stream>>>(dlt, rto, en, (float*)d_out);
}
